// HyperCDM_EX_71700184039602
// MI455X (gfx1250) — matrix operand order repaired and checked by reading, not compiled, not run on hardware
//
#include <hip/hip_runtime.h>
#include <hip/hip_bf16.h>
#include <hip/hip_fp16.h>

#if __has_include(<hip/amd_detail/amd_gfx1250_TDM.h>)
#define TDM_6ARG 1
#endif

// ---- problem sizes (match reference) ----
#define S_N   100000
#define E_N   30000
#define K_N   1000
#define D_DIM 64
#define F_DIM 128
#define B_N   16384
#define NNZ_S 1600000
#define NNZ_E 480000
#define NNZ_K 32000
#define LEAK  0.1f
#define MOM   0.9f

typedef _Float16 v16h __attribute__((ext_vector_type(16)));
typedef _Float16 v8h  __attribute__((ext_vector_type(8)));
typedef float    v8f  __attribute__((ext_vector_type(8)));
typedef unsigned int u32x4 __attribute__((ext_vector_type(4)));
typedef int          i32x4 __attribute__((ext_vector_type(4)));
typedef int          i32x8 __attribute__((ext_vector_type(8)));

#define CDIV(a, b) (((a) + (b) - 1) / (b))

// ---------------- elementwise / conv kernels (f32, memory-bound) ----------------
__global__ void k_copy(float* __restrict__ dst, const float* __restrict__ src, int n) {
    int i = blockIdx.x * blockDim.x + threadIdx.x;
    if (i < n) dst[i] = src[i];
}

__global__ void k_scale(float* __restrict__ dst, const float* __restrict__ src, float s, int n) {
    int i = blockIdx.x * blockDim.x + threadIdx.x;
    if (i < n) dst[i] = s * src[i];
}

__global__ void k_add(float* __restrict__ dst, const float* __restrict__ src, int n) {
    int i = blockIdx.x * blockDim.x + threadIdx.x;
    if (i < n) dst[i] += src[i];
}

// COO SpMM scatter: y[row,:] += val * x[col,:]; one thread per (nnz, dim).
__global__ void k_spmm(const int* __restrict__ rows, const int* __restrict__ cols,
                       const float* __restrict__ vals, const float* __restrict__ x,
                       float* __restrict__ y, int nnz) {
    long long idx = (long long)blockIdx.x * blockDim.x + threadIdx.x;
    int e = (int)(idx >> 6);
    int d = (int)(idx & 63);
    if (e < nnz) {
        int r = rows[e], c = cols[e];
        atomicAdd(&y[(long long)r * D_DIM + d], vals[e] * x[(long long)c * D_DIM + d]);
    }
}

__global__ void k_f2h(_Float16* __restrict__ dst, const float* __restrict__ src, int n) {
    int i = blockIdx.x * blockDim.x + threadIdx.x;
    if (i < n) dst[i] = (_Float16)src[i];
}

__global__ void k_gather_h(_Float16* __restrict__ dst, const float* __restrict__ src,
                           const int* __restrict__ ids, int bcount) {
    int i = blockIdx.x * blockDim.x + threadIdx.x;
    if (i < bcount * D_DIM) {
        int b = i >> 6, d = i & 63;
        dst[i] = (_Float16)src[(long long)ids[b] * D_DIM + d];
    }
}

__global__ void k_gather_f(float* __restrict__ dst, const float* __restrict__ src,
                           const int* __restrict__ ids, int bcount) {
    int i = blockIdx.x * blockDim.x + threadIdx.x;
    if (i < bcount * D_DIM) {
        int b = i >> 6, d = i & 63;
        dst[i] = src[(long long)ids[b] * D_DIM + d];
    }
}

// ---------------- Tensor Data Mover: 2D f16 tile -> LDS ----------------
// D# packing per CDNA5 ISA (2-D tensor, groups 2/3 zero).
__device__ __forceinline__ void tdm_load_tile_f16(unsigned lds_off, const _Float16* gsrc,
                                                  unsigned tensor_d0, unsigned tensor_d1,
                                                  unsigned tile_d0, unsigned tile_d1,
                                                  unsigned long long d0_stride) {
    unsigned long long ga = (unsigned long long)(size_t)gsrc;
    u32x4 g0;
    g0[0] = 1u;                                              // count=1, user mode
    g0[1] = lds_off;                                         // lds_addr (bytes)
    g0[2] = (unsigned)(ga & 0xFFFFFFFFu);                    // global_addr[31:0]
    g0[3] = (unsigned)((ga >> 32) & 0x01FFFFFFu) | (2u << 30); // addr[56:32] | type=2
    i32x8 g1;
    g1[0] = (int)(1u << 16);                                 // data_size=1 -> 2 bytes; wg_mask=0
    g1[1] = (int)((tensor_d0 & 0xFFFFu) << 16);              // tensor_dim0[15:0] @ bits 63:48
    g1[2] = (int)((tensor_d0 >> 16) | ((tensor_d1 & 0xFFFFu) << 16)); // td0 hi | td1 lo
    g1[3] = (int)((tensor_d1 >> 16) | (tile_d0 << 16));      // td1 hi | tile_dim0
    g1[4] = (int)(tile_d1 & 0xFFFFu);                        // tile_dim1 | tile_dim2=0
    g1[5] = (int)(d0_stride & 0xFFFFFFFFull);                // tensor_dim0_stride[31:0]
    g1[6] = (int)((d0_stride >> 32) & 0xFFFFull);            // stride[47:32] | dim1_stride=0
    g1[7] = 0;
    i32x4 z4 = {0, 0, 0, 0};
#if defined(TDM_6ARG)
    i32x8 z8 = {0, 0, 0, 0, 0, 0, 0, 0};
    __builtin_amdgcn_tensor_load_to_lds(g0, g1, z4, z4, z8, 0);
#else
    __builtin_amdgcn_tensor_load_to_lds(g0, g1, z4, z4, 0);
#endif
}

// ---------------- WMMA GEMM kernels (f16 in, f32 accumulate) ----------------
// Fragment mapping used by these kernels (wave32; contiguous k, same for A and B):
//   lane: n = lane&15, hi = lane>>4
//   A (16x32 f16, v16h): row = n;  k(j) = hi*16 + j
//   B (32x16 f16, v16h): col = n;  k(j) = hi*16 + j
//   C/D (16x16 f32, v8f): row = hi*8 + j; col = n

// Y[M,128] = leaky_relu(X[M,64] @ W[64,128] + bias); one wave = one 16x16 tile.
__global__ void k_linear_wmma(const _Float16* __restrict__ X, const _Float16* __restrict__ W,
                              const float* __restrict__ bias, _Float16* __restrict__ outH,
                              float* __restrict__ outF, int M) {
    int lane = threadIdx.x;
    int n = lane & 15, hi = lane >> 4;
    int tm = blockIdx.x, tn = blockIdx.y;  // tn in [0,8): 8 tiles cover F=128
    int col = tn * 16 + n;
    int r = tm * 16 + n;
    if (r >= M) r = M - 1;  // clamp (keeps EXEC all-ones for WMMA)

    v8f c = {};
#pragma unroll
    for (int kb = 0; kb < D_DIM; kb += 32) {
        const _Float16* pa = X + (long long)r * D_DIM + kb + hi * 16;
        v8h alo = *(const v8h*)pa;
        v8h ahi = *(const v8h*)(pa + 8);
        v16h a;
#pragma unroll
        for (int j = 0; j < 8; ++j) { a[j] = alo[j]; a[8 + j] = ahi[j]; }
        v16h b;
#pragma unroll
        for (int j = 0; j < 16; ++j) b[j] = W[(kb + hi * 16 + j) * F_DIM + col];
        c = __builtin_amdgcn_wmma_f32_16x16x32_f16(false, a, false, b, (short)0, c, false, false);
    }
    float bv = bias[col];
#pragma unroll
    for (int j = 0; j < 8; ++j) {
        int row = tm * 16 + hi * 8 + j;
        if (row < M) {
            float v = c[j] + bv;
            v = (v > 0.0f) ? v : LEAK * v;
            outH[(long long)row * F_DIM + col] = (_Float16)v;
            if (outF) outF[(long long)row * F_DIM + col] = v;
        }
    }
}

// O[B,1000] = A[B,128] @ Kn[1000,128]^T.
// Block = 8 waves: TDM-stage A tile (16x128) + Kn stripe (128x128) into LDS,
// then each wave computes one 16x16 tile from LDS. blockIdx.z selects stream.
__global__ void __launch_bounds__(256)
k_bigmm_wmma(const _Float16* __restrict__ A0, const _Float16* __restrict__ A1,
             const _Float16* __restrict__ Kn,
             float* __restrict__ O0, float* __restrict__ O1) {
    __shared__ _Float16 sA[16 * F_DIM];    // 4 KB
    __shared__ _Float16 sK[128 * F_DIM];   // 32 KB

    const _Float16* A = blockIdx.z ? A1 : A0;
    float* O = blockIdx.z ? O1 : O0;
    int tm = blockIdx.x;        // M tile (16 rows)
    int tnb = blockIdx.y;       // N stripe (128 cols of K_N)

    if (threadIdx.x < 32) {     // wave 0 issues both DMAs (TDM ignores EXEC)
        unsigned offA = (unsigned)(size_t)(&sA[0]);
        unsigned offK = (unsigned)(size_t)(&sK[0]);
        // A tile: rows tm*16 .. +16, all 128 k
        tdm_load_tile_f16(offA, A + (long long)tm * 16 * F_DIM,
                          F_DIM, 16, F_DIM, 16, F_DIM);
        // Kn stripe: rows tnb*128 .. +128 (OOB rows zero-filled via tensor_dim1)
        unsigned remain = (unsigned)(K_N - tnb * 128);
        tdm_load_tile_f16(offK, Kn + (long long)tnb * 128 * F_DIM,
                          F_DIM, remain, F_DIM, 128, F_DIM);
        __builtin_amdgcn_s_wait_tensorcnt(0);
    }
    __syncthreads();

    int lane = threadIdx.x & 31;
    int w = threadIdx.x >> 5;   // wave -> 16-col subtile within the stripe
    int n = lane & 15, hi = lane >> 4;

    v8f c = {};
#pragma unroll
    for (int kb = 0; kb < F_DIM; kb += 32) {
        const _Float16* pa = &sA[n * F_DIM + kb + hi * 16];
        v8h alo = *(const v8h*)pa;
        v8h ahi = *(const v8h*)(pa + 8);
        v16h a;
#pragma unroll
        for (int j = 0; j < 8; ++j) { a[j] = alo[j]; a[8 + j] = ahi[j]; }
        const _Float16* pb = &sK[(w * 16 + n) * F_DIM + kb + hi * 16];
        v8h blo = *(const v8h*)pb;
        v8h bhi = *(const v8h*)(pb + 8);
        v16h b;
#pragma unroll
        for (int j = 0; j < 8; ++j) { b[j] = blo[j]; b[8 + j] = bhi[j]; }
        c = __builtin_amdgcn_wmma_f32_16x16x32_f16(false, a, false, b, (short)0, c, false, false);
    }

    int ncol = tnb * 128 + w * 16 + n;
#pragma unroll
    for (int j = 0; j < 8; ++j) {
        int row = tm * 16 + hi * 8 + j;
        if (ncol < K_N) O[(long long)row * K_N + ncol] = c[j];
    }
}

// disc = sigmoid(batch_exercise @ W_disc + b_disc)
__global__ void k_disc(const float* __restrict__ exf, const int* __restrict__ eid,
                       const float* __restrict__ Wd, const float* __restrict__ bd,
                       float* __restrict__ out) {
    int b = blockIdx.x * blockDim.x + threadIdx.x;
    if (b < B_N) {
        const float* x = exf + (long long)eid[b] * D_DIM;
        float s = bd[0];
#pragma unroll
        for (int d = 0; d < D_DIM; ++d) s += x[d] * Wd[d];
        out[b] = 1.0f / (1.0f + __expf(-s));
    }
}

// ---------------- host orchestration ----------------
extern "C" void kernel_launch(void* const* d_in, const int* in_sizes, int n_in,
                              void* d_out, int out_size, void* d_ws, size_t ws_size,
                              hipStream_t stream) {
    const int*   student_id  = (const int*)d_in[0];
    const int*   exercise_id = (const int*)d_in[1];
    // d_in[2] = q_mask (unused by outputs)
    const float* stu_emb    = (const float*)d_in[3];
    const float* exer_emb   = (const float*)d_in[4];
    const float* know_emb   = (const float*)d_in[5];
    const float* impact_emb = (const float*)d_in[6];
    const int*   s_rows = (const int*)d_in[7];
    const int*   s_cols = (const int*)d_in[8];
    const float* s_vals = (const float*)d_in[9];
    const int*   e_rows = (const int*)d_in[10];
    const int*   e_cols = (const int*)d_in[11];
    const float* e_vals = (const float*)d_in[12];
    const int*   k_rows = (const int*)d_in[13];
    const int*   k_cols = (const int*)d_in[14];
    const float* k_vals = (const float*)d_in[15];
    const float* W_stu  = (const float*)d_in[16];
    const float* b_stu  = (const float*)d_in[17];
    const float* W_exer = (const float*)d_in[18];
    const float* b_exer = (const float*)d_in[19];
    const float* W_know = (const float*)d_in[20];
    const float* b_know = (const float*)d_in[21];
    const float* W_disc = (const float*)d_in[22];
    const float* b_disc = (const float*)d_in[23];
    float* out = (float*)d_out;

    // output layout (flat concat, reference return order)
    float* out_student = out;                                   // [B, K]
    float* out_diff    = out + (size_t)B_N * K_N;               // [B, K]
    float* out_disc    = out + 2ull * B_N * K_N;                // [B, 1]
    float* out_kts     = out + 2ull * B_N * K_N + B_N;          // [K, F]
    float* out_imp     = out_kts + (size_t)K_N * F_DIM;         // [B, D]

    // workspace carve-up (16B-aligned chunks)
    char* ws = (char*)d_ws;
    size_t off = 0;
    auto allocF = [&](size_t nf) -> float* {
        float* p = (float*)(ws + off); off += ((nf * 4 + 15) / 16) * 16; return p;
    };
    auto allocH = [&](size_t nh) -> _Float16* {
        _Float16* p = (_Float16*)(ws + off); off += ((nh * 2 + 15) / 16) * 16; return p;
    };

    float* sc = allocF((size_t)S_N * D_DIM);
    float* sn = allocF((size_t)S_N * D_DIM);
    float* sa = allocF((size_t)S_N * D_DIM);
    float* ec = allocF((size_t)E_N * D_DIM);
    float* en = allocF((size_t)E_N * D_DIM);
    float* ea = allocF((size_t)E_N * D_DIM);
    float* kc = allocF((size_t)K_N * D_DIM);
    float* kx = allocF((size_t)K_N * D_DIM);
    float* ka = allocF((size_t)K_N * D_DIM);
    _Float16* bsH   = allocH((size_t)B_N * D_DIM);
    _Float16* beH   = allocH((size_t)B_N * D_DIM);
    _Float16* knInH = allocH((size_t)K_N * D_DIM);
    _Float16* stH   = allocH((size_t)B_N * F_DIM);
    _Float16* dfH   = allocH((size_t)B_N * F_DIM);
    _Float16* ktsH  = allocH((size_t)K_N * F_DIM);
    _Float16* WsH   = allocH((size_t)D_DIM * F_DIM);
    _Float16* WeH   = allocH((size_t)D_DIM * F_DIM);
    _Float16* WkH   = allocH((size_t)D_DIM * F_DIM);

    const int TB = 256;
    // conv: acc=emb; cur=emb; 3x { nxt = spmm(cur) + 0.9*cur; acc += nxt; swap }; acc *= 0.25
    auto conv = [&](const float* emb, const int* rr, const int* cc, const float* vv,
                    int nnz, int nrows, float* cur, float* nxt, float* acc) {
        int nd = nrows * D_DIM;
        k_copy<<<CDIV(nd, TB), TB, 0, stream>>>(cur, emb, nd);
        k_copy<<<CDIV(nd, TB), TB, 0, stream>>>(acc, emb, nd);
        float* cu = cur; float* nx = nxt;
        for (int l = 0; l < 3; ++l) {
            k_scale<<<CDIV(nd, TB), TB, 0, stream>>>(nx, cu, MOM, nd);
            long long work = (long long)nnz * D_DIM;
            k_spmm<<<(unsigned)CDIV(work, TB), TB, 0, stream>>>(rr, cc, vv, cu, nx, nnz);
            k_add<<<CDIV(nd, TB), TB, 0, stream>>>(acc, nx, nd);
            float* t = cu; cu = nx; nx = t;
        }
        k_scale<<<CDIV(nd, TB), TB, 0, stream>>>(acc, acc, 0.25f, nd);
    };

    conv(stu_emb,  s_rows, s_cols, s_vals, NNZ_S, S_N, sc, sn, sa);
    conv(exer_emb, e_rows, e_cols, e_vals, NNZ_E, E_N, ec, en, ea);
    conv(know_emb, k_rows, k_cols, k_vals, NNZ_K, K_N, kc, kx, ka);

    // weight + feature conversions to f16 (WMMA operand precision)
    k_f2h<<<CDIV(D_DIM * F_DIM, TB), TB, 0, stream>>>(WsH, W_stu, D_DIM * F_DIM);
    k_f2h<<<CDIV(D_DIM * F_DIM, TB), TB, 0, stream>>>(WeH, W_exer, D_DIM * F_DIM);
    k_f2h<<<CDIV(D_DIM * F_DIM, TB), TB, 0, stream>>>(WkH, W_know, D_DIM * F_DIM);
    k_gather_h<<<CDIV(B_N * D_DIM, TB), TB, 0, stream>>>(bsH, sa, student_id, B_N);
    k_gather_h<<<CDIV(B_N * D_DIM, TB), TB, 0, stream>>>(beH, ea, exercise_id, B_N);
    k_f2h<<<CDIV(K_N * D_DIM, TB), TB, 0, stream>>>(knInH, ka, K_N * D_DIM);
    k_gather_f<<<CDIV(B_N * D_DIM, TB), TB, 0, stream>>>(out_imp, impact_emb, exercise_id, B_N);

    // feature linears (WMMA): one wave per 16x16 output tile
    dim3 glB(B_N / 16, F_DIM / 16);
    k_linear_wmma<<<glB, 32, 0, stream>>>(bsH, WsH, b_stu, stH, nullptr, B_N);
    k_linear_wmma<<<glB, 32, 0, stream>>>(beH, WeH, b_exer, dfH, nullptr, B_N);
    dim3 glK(CDIV(K_N, 16), F_DIM / 16);
    k_linear_wmma<<<glK, 32, 0, stream>>>(knInH, WkH, b_know, ktsH, out_kts, K_N);

    // big GEMMs: TDM-staged LDS tiles, 8 waves/block
    dim3 gbg(B_N / 16, CDIV(K_N, 128), 2);
    k_bigmm_wmma<<<gbg, 256, 0, stream>>>(stH, dfH, ktsH, out_student, out_diff);

    // disc head
    k_disc<<<CDIV(B_N, TB), TB, 0, stream>>>(ea, exercise_id, W_disc, b_disc, out_disc);
}
